// Attention_33389075759323
// MI455X (gfx1250) — matrix operand order repaired and checked by reading, not compiled, not run on hardware
//
#include <hip/hip_runtime.h>

// ---------------- CDNA5 WMMA types ----------------
typedef __attribute__((ext_vector_type(16))) __bf16 v16bf;
typedef __attribute__((ext_vector_type(8)))  __bf16 v8bf;
typedef __attribute__((ext_vector_type(8)))  float  v8f;
typedef __attribute__((ext_vector_type(4)))  float  v4f;
typedef __attribute__((ext_vector_type(4)))  unsigned int v4u;

#define DIM      512
#define HEADS    8
#define DH       64
#define INNER    512
#define NQKV     1536
#define SEQ      4096
#define BATCH    2
#define MROWS    (BATCH * SEQ)   // 8192
#define SCALE_Q  0.125f          // 64^-0.5

__device__ __forceinline__ v8f wmma_bf16(v16bf a, v16bf b, v8f c) {
  return __builtin_amdgcn_wmma_f32_16x16x32_bf16(
      /*neg_a=*/false, a, /*neg_b=*/false, b,
      /*c_mod=*/(short)0, c, /*reuse_a=*/false, /*reuse_b=*/false);
}

// A-matrix 16x32 bf16 (ISA 7.12.2): per lane two contiguous 16B runs:
// K = half*8..+7 and K = 16+half*8..+7. ld % 8 == 0, 16B-aligned base.
__device__ __forceinline__ v16bf load_a_frag(const __bf16* base, int ld) {
  const int lane = threadIdx.x & 31;
  const int m = lane & 15, half = lane >> 4;
  const __bf16* p = base + m * ld + half * 8;
  const v8bf lo = *(const v8bf*)(p);
  const v8bf hi = *(const v8bf*)(p + 16);
  return __builtin_shufflevector(lo, hi, 0,1,2,3,4,5,6,7,8,9,10,11,12,13,14,15);
}

// B-matrix 32x16 where memory holds N x K row-major (B[k,n] = base[n*ld + k]).
// Per lane: K = khalf*8..+7 and K = 16+khalf*8..+7 -> 2x b128. ld % 8 == 0.
__device__ __forceinline__ v16bf load_bT_frag(const __bf16* base, int ld) {
  const int lane = threadIdx.x & 31;
  const int n = lane & 15, khalf = lane >> 4;
  const __bf16* p = base + n * ld + khalf * 8;
  const v8bf lo = *(const v8bf*)(p);
  const v8bf hi = *(const v8bf*)(p + 16);
  return __builtin_shufflevector(lo, hi, 0,1,2,3,4,5,6,7,8,9,10,11,12,13,14,15);
}

__device__ __forceinline__ v8bf cvt8(v4f f0, v4f f1) {
  v8bf o;
#pragma unroll
  for (int j = 0; j < 4; ++j) { o[j] = (__bf16)f0[j]; o[4 + j] = (__bf16)f1[j]; }
  return o;
}

// ---- gfx1250 async global->LDS copy (ASYNCcnt-tracked, no VGPR round trip) ----
__device__ __forceinline__ unsigned lds_off32(const void* p) {
  // AS3 offsets occupy the low 32 bits of a generic pointer on AMDGPU.
  return (unsigned)(unsigned long long)(__UINTPTR_TYPE__)p;
}
__device__ __forceinline__ void async_load_b128(unsigned ldsoff, const void* gaddr) {
  asm volatile("global_load_async_to_lds_b128 %0, %1, off"
               :: "v"(ldsoff), "v"(gaddr) : "memory");
}
__device__ __forceinline__ void wait_async0() {
  asm volatile("s_wait_asynccnt 0x0" ::: "memory");
}

// ---------------- Kernel 1: QKV projection ----------------
// C[8192x1536] = x[8192x512] @ Wqkv[512x1536]; scatter into q/k/v [bh][l][dh].
// 64x64 block tile, K-chunk 64: 4 WMMAs per wave between one barrier pair.
__global__ __launch_bounds__(256)
void qkv_gemm_kernel(const float* __restrict__ x, const float* __restrict__ Wqkv,
                     __bf16* __restrict__ qb, __bf16* __restrict__ kb, __bf16* __restrict__ vb) {
  __shared__ __attribute__((aligned(16))) __bf16 As[64 * 72];  // 64 M x 64 K, ld 72
  __shared__ __attribute__((aligned(16))) __bf16 Bs[64 * 72];  // 64 N x 64 K (transposed), ld 72
  const int t = threadIdx.x, wid = t >> 5, lane = t & 31;
  const int m0 = blockIdx.x * 64, n0 = blockIdx.y * 64;
  const int mt = wid >> 2, nt = wid & 3;
  const int ar = t >> 2, ac = (t & 3) * 16;       // A: row, 16-elem segment
  const int bn = t & 63, bk = (t >> 6) * 16;      // B: N col, 16 K-rows
  v8f acc0 = {}, acc1 = {};
  for (int k0 = 0; k0 < DIM; k0 += 64) {
    {  // A stage: 4x b128 fp32 loads -> convert -> 2x b128 stores
      const float* xp = x + (size_t)(m0 + ar) * DIM + k0 + ac;
      const v4f f0 = *(const v4f*)(xp);
      const v4f f1 = *(const v4f*)(xp + 4);
      const v4f f2 = *(const v4f*)(xp + 8);
      const v4f f3 = *(const v4f*)(xp + 12);
      *(v8bf*)(&As[ar * 72 + ac]) = cvt8(f0, f1);
      *(v8bf*)(&As[ar * 72 + ac + 8]) = cvt8(f2, f3);
    }
    {  // B stage (transpose): 16 coalesced b32 loads -> 2x b128 stores (N-major)
      float w[16];
#pragma unroll
      for (int j = 0; j < 16; ++j)
        w[j] = Wqkv[(size_t)(k0 + bk + j) * NQKV + n0 + bn];
      v8bf o0, o1;
#pragma unroll
      for (int j = 0; j < 8; ++j) { o0[j] = (__bf16)w[j]; o1[j] = (__bf16)w[8 + j]; }
      *(v8bf*)(&Bs[bn * 72 + bk]) = o0;
      *(v8bf*)(&Bs[bn * 72 + bk + 8]) = o1;
    }
    __syncthreads();
#pragma unroll
    for (int ks = 0; ks < 64; ks += 32) {
      const v16bf b  = load_bT_frag(&Bs[(nt * 16) * 72 + ks], 72);
      const v16bf a0 = load_a_frag(&As[(mt * 32) * 72 + ks], 72);
      const v16bf a1 = load_a_frag(&As[(mt * 32 + 16) * 72 + ks], 72);
      acc0 = wmma_bf16(a0, b, acc0);
      acc1 = wmma_bf16(a1, b, acc1);
    }
    __syncthreads();
  }
  // Scatter epilogue into head-split layout; fold softmax scale into q.
  const int n = lane & 15, half = lane >> 4;
  const int gcol  = n0 + nt * 16 + n;
  const int which = gcol / INNER;          // 0=q 1=k 2=v
  const int inner = gcol % INNER;
  const int h = inner / DH, d = inner % DH;
  __bf16* dst = (which == 0) ? qb : ((which == 1) ? kb : vb);
  const float scl = (which == 0) ? SCALE_Q : 1.0f;
#pragma unroll
  for (int tile = 0; tile < 2; ++tile) {
    const v8f& acc = tile ? acc1 : acc0;
#pragma unroll
    for (int r = 0; r < 8; ++r) {
      const int gm = m0 + mt * 32 + tile * 16 + r + half * 8;
      const int bidx = gm >> 12, l = gm & (SEQ - 1);
      dst[(((size_t)(bidx * HEADS + h)) * SEQ + l) * DH + d] = (__bf16)(acc[r] * scl);
    }
  }
}

// ---------------- Kernel 2: flash attention ----------------
// Grid: (bh=16, rowblocks=32). 8 waves/block, each wave: 16 query rows x 64 dims.
__global__ __launch_bounds__(256)
void flash_attn_kernel(const __bf16* __restrict__ qb, const __bf16* __restrict__ kb,
                       const __bf16* __restrict__ vb, __bf16* __restrict__ attned) {
  __shared__ __attribute__((aligned(16))) __bf16 ldsK[64 * 72];   // [key][d], ld 72
  __shared__ __attribute__((aligned(16))) __bf16 ldsVt[64 * 72];  // [d][key], ld 72
  __shared__ __attribute__((aligned(16))) __bf16 pbuf[8][16 * 72];
  const int t = threadIdx.x, wid = t >> 5, lane = t & 31;
  const int half = lane >> 4;
  const int bh = blockIdx.x;
  const int q0 = blockIdx.y * 128 + wid * 16;

  const __bf16* qbase = qb + ((size_t)bh * SEQ + q0) * DH;
  const __bf16* kbase = kb + (size_t)bh * SEQ * DH;
  const __bf16* vbase = vb + (size_t)bh * SEQ * DH;

  // Q fragments resident in registers (2x global_load_b128 each).
  const v16bf aq0 = load_a_frag(qbase, DH);        // d = 0..31
  const v16bf aq1 = load_a_frag(qbase + 32, DH);   // d = 32..63

  v8f o0 = {}, o1 = {}, o2 = {}, o3 = {};
  float mrow[8], lrow[8];
#pragma unroll
  for (int r = 0; r < 8; ++r) { mrow[r] = -1e30f; lrow[r] = 0.f; }

  // fixed staging slices
  const int kr = t >> 2, kc = (t & 3) * 8;     // K: key row, 8-dword segment
  const int vd = (t & 31) * 2;                 // V: d pair
  const int vk = (t >> 5) * 8;                 // V: 8 keys
  const unsigned ldsk_off0 = lds_off32(&ldsK[kr * 72 + kc * 2]);
  const unsigned ldsk_off1 = lds_off32(&ldsK[kr * 72 + kc * 2 + 8]);

  for (int kc0 = 0; kc0 < SEQ; kc0 += 64) {
    // ---- stage K via async global->LDS b128 (ASYNCcnt path) ----
    {
      const unsigned int* gk = (const unsigned int*)(kbase + (size_t)kc0 * DH);
      async_load_b128(ldsk_off0, gk + kr * 32 + kc);
      async_load_b128(ldsk_off1, gk + kr * 32 + kc + 4);
    }
    // ---- stage V transposed: 8 coalesced b32 loads -> 2 b128 stores ----
    {
      const unsigned int* gv = (const unsigned int*)(vbase + (size_t)kc0 * DH);
      unsigned int w[8];
#pragma unroll
      for (int j = 0; j < 8; ++j) w[j] = gv[(vk + j) * 32 + (t & 31)];
      v8bf lo, hi;
#pragma unroll
      for (int j = 0; j < 8; ++j) {
        lo[j] = __builtin_bit_cast(__bf16, (unsigned short)(w[j] & 0xffffu));
        hi[j] = __builtin_bit_cast(__bf16, (unsigned short)(w[j] >> 16));
      }
      *(v8bf*)(&ldsVt[vd * 72 + vk]) = lo;
      *(v8bf*)(&ldsVt[(vd + 1) * 72 + vk]) = hi;
    }
    if (kc0 + 64 < SEQ) {  // prefetch next chunk (global_prefetch_b8)
      __builtin_prefetch(kbase + (size_t)(kc0 + 64) * DH + t * 16, 0, 3);
      __builtin_prefetch(vbase + (size_t)(kc0 + 64) * DH + t * 16, 0, 3);
    }
    wait_async0();
    __syncthreads();

    // ---- S = Q K^T : four 16x16 tiles over 64 keys ----
    v8f s0 = {}, s1 = {}, s2 = {}, s3 = {};
    s0 = wmma_bf16(aq0, load_bT_frag(&ldsK[0 * 72], 72), s0);
    s0 = wmma_bf16(aq1, load_bT_frag(&ldsK[0 * 72 + 32], 72), s0);
    s1 = wmma_bf16(aq0, load_bT_frag(&ldsK[16 * 72], 72), s1);
    s1 = wmma_bf16(aq1, load_bT_frag(&ldsK[16 * 72 + 32], 72), s1);
    s2 = wmma_bf16(aq0, load_bT_frag(&ldsK[32 * 72], 72), s2);
    s2 = wmma_bf16(aq1, load_bT_frag(&ldsK[32 * 72 + 32], 72), s2);
    s3 = wmma_bf16(aq0, load_bT_frag(&ldsK[48 * 72], 72), s3);
    s3 = wmma_bf16(aq1, load_bT_frag(&ldsK[48 * 72 + 32], 72), s3);

    // ---- online softmax; C-layout: vgpr r -> row m = r + half*8, col n = lane&15
    __bf16* pw = &pbuf[wid][0];
    const int n = lane & 15;
#pragma unroll
    for (int r = 0; r < 8; ++r) {
      const float v0 = s0[r], v1 = s1[r], v2 = s2[r], v3 = s3[r];
      float mx = fmaxf(fmaxf(v0, v1), fmaxf(v2, v3));
#pragma unroll
      for (int off = 1; off < 16; off <<= 1) mx = fmaxf(mx, __shfl_xor(mx, off, 16));
      const float mnew  = fmaxf(mrow[r], mx);
      const float alpha = __expf(mrow[r] - mnew);
      const float p0 = __expf(v0 - mnew);
      const float p1 = __expf(v1 - mnew);
      const float p2 = __expf(v2 - mnew);
      const float p3 = __expf(v3 - mnew);
      float ps = (p0 + p1) + (p2 + p3);
#pragma unroll
      for (int off = 1; off < 16; off <<= 1) ps += __shfl_xor(ps, off, 16);
      lrow[r] = lrow[r] * alpha + ps;
      mrow[r] = mnew;
      o0[r] *= alpha; o1[r] *= alpha; o2[r] *= alpha; o3[r] *= alpha;
      const int m = r + half * 8;
      pw[m * 72 + n]      = (__bf16)p0;
      pw[m * 72 + 16 + n] = (__bf16)p1;
      pw[m * 72 + 32 + n] = (__bf16)p2;
      pw[m * 72 + 48 + n] = (__bf16)p3;
    }

    // ---- O += P(16x64) * V(64x64), V read d-major (wide bT loads) ----
    const v16bf pf0 = load_a_frag(pw, 72);        // keys 0..31
    const v16bf pf1 = load_a_frag(pw + 32, 72);   // keys 32..63
    o0 = wmma_bf16(pf0, load_bT_frag(&ldsVt[0 * 72], 72), o0);
    o0 = wmma_bf16(pf1, load_bT_frag(&ldsVt[0 * 72 + 32], 72), o0);
    o1 = wmma_bf16(pf0, load_bT_frag(&ldsVt[16 * 72], 72), o1);
    o1 = wmma_bf16(pf1, load_bT_frag(&ldsVt[16 * 72 + 32], 72), o1);
    o2 = wmma_bf16(pf0, load_bT_frag(&ldsVt[32 * 72], 72), o2);
    o2 = wmma_bf16(pf1, load_bT_frag(&ldsVt[32 * 72 + 32], 72), o2);
    o3 = wmma_bf16(pf0, load_bT_frag(&ldsVt[48 * 72], 72), o3);
    o3 = wmma_bf16(pf1, load_bT_frag(&ldsVt[48 * 72 + 32], 72), o3);
    __syncthreads();
  }

  // epilogue: attned [b, l, h*64+d] bf16
  const int bidx = bh >> 3, h = bh & 7;
  const int n = lane & 15;
#pragma unroll
  for (int r = 0; r < 8; ++r) {
    const float inv = 1.0f / lrow[r];
    const int l = q0 + r + half * 8;
    const size_t base = ((size_t)(bidx * SEQ + l)) * DIM + h * DH;
    attned[base + 0  + n] = (__bf16)(o0[r] * inv);
    attned[base + 16 + n] = (__bf16)(o1[r] * inv);
    attned[base + 32 + n] = (__bf16)(o2[r] * inv);
    attned[base + 48 + n] = (__bf16)(o3[r] * inv);
  }
}

// ---------------- Kernel 3: output projection + bias ----------------
__global__ __launch_bounds__(256)
void out_gemm_kernel(const __bf16* __restrict__ A, const float* __restrict__ Wout,
                     const float* __restrict__ bias, float* __restrict__ out) {
  __shared__ __attribute__((aligned(16))) __bf16 As[64 * 72];
  __shared__ __attribute__((aligned(16))) __bf16 Bs[64 * 72];  // transposed N x K
  const int t = threadIdx.x, wid = t >> 5, lane = t & 31;
  const int m0 = blockIdx.x * 64, n0 = blockIdx.y * 64;
  const int mt = wid >> 2, nt = wid & 3;
  const int ar = t >> 2, ac = (t & 3) * 16;
  const int bn = t & 63, bk = (t >> 6) * 16;
  v8f acc0 = {}, acc1 = {};
  for (int k0 = 0; k0 < INNER; k0 += 64) {
    {  // A stage: bf16 source, 2x b128 load + 2x b128 store
      const __bf16* ap = A + (size_t)(m0 + ar) * INNER + k0 + ac;
      *(v8bf*)(&As[ar * 72 + ac])     = *(const v8bf*)(ap);
      *(v8bf*)(&As[ar * 72 + ac + 8]) = *(const v8bf*)(ap + 8);
    }
    {  // B stage (transpose): 16 coalesced b32 loads -> 2x b128 stores
      float w[16];
#pragma unroll
      for (int j = 0; j < 16; ++j)
        w[j] = Wout[(size_t)(k0 + bk + j) * DIM + n0 + bn];
      v8bf o0, o1;
#pragma unroll
      for (int j = 0; j < 8; ++j) { o0[j] = (__bf16)w[j]; o1[j] = (__bf16)w[8 + j]; }
      *(v8bf*)(&Bs[bn * 72 + bk]) = o0;
      *(v8bf*)(&Bs[bn * 72 + bk + 8]) = o1;
    }
    __syncthreads();
#pragma unroll
    for (int ks = 0; ks < 64; ks += 32) {
      const v16bf b  = load_bT_frag(&Bs[(nt * 16) * 72 + ks], 72);
      const v16bf a0 = load_a_frag(&As[(mt * 32) * 72 + ks], 72);
      const v16bf a1 = load_a_frag(&As[(mt * 32 + 16) * 72 + ks], 72);
      acc0 = wmma_bf16(a0, b, acc0);
      acc1 = wmma_bf16(a1, b, acc1);
    }
    __syncthreads();
  }
  const int n = lane & 15, half = lane >> 4;
  const int gcol = n0 + nt * 16 + n;
  const float bv = bias[gcol];
#pragma unroll
  for (int tile = 0; tile < 2; ++tile) {
    const v8f& acc = tile ? acc1 : acc0;
#pragma unroll
    for (int r = 0; r < 8; ++r) {
      const int gm = m0 + mt * 32 + tile * 16 + r + half * 8;
      out[(size_t)gm * DIM + gcol] = acc[r] + bv;
    }
  }
}

// ---------------- launch ----------------
extern "C" void kernel_launch(void* const* d_in, const int* in_sizes, int n_in,
                              void* d_out, int out_size, void* d_ws, size_t ws_size,
                              hipStream_t stream) {
  const float* x    = (const float*)d_in[0];
  const float* Wqkv = (const float*)d_in[1];
  const float* Wout = (const float*)d_in[2];
  const float* bout = (const float*)d_in[3];
  float* out = (float*)d_out;

  char* ws = (char*)d_ws;
  const size_t qbytes = (size_t)BATCH * HEADS * SEQ * DH * sizeof(__bf16); // 8 MB
  __bf16* qb     = (__bf16*)(ws);
  __bf16* kb     = (__bf16*)(ws + qbytes);
  __bf16* vb     = (__bf16*)(ws + 2 * qbytes);
  __bf16* attned = (__bf16*)(ws + 3 * qbytes);

  dim3 g1(MROWS / 64, NQKV / 64);    // 128 x 24
  qkv_gemm_kernel<<<g1, 256, 0, stream>>>(x, Wqkv, qb, kb, vb);

  dim3 g2(BATCH * HEADS, SEQ / 128); // 16 x 32
  flash_attn_kernel<<<g2, 256, 0, stream>>>(qb, kb, vb, attned);

  dim3 g3(MROWS / 64, DIM / 64);     // 128 x 8
  out_gemm_kernel<<<g3, 256, 0, stream>>>(attned, Wout, bout, out);
}
